// BahdanauAttention_61624190763471
// MI455X (gfx1250) — matrix operand order repaired and checked by reading, not compiled, not run on hardware
//
#include <hip/hip_runtime.h>

// ---- problem constants ----
#define B_   16
#define L_   2048
#define DF_  512
#define H_   8
#define DH_  512
#define U_   512

// ---- score-GEMM tiling ----
#define TM   64          // rows per block
#define KPA  520         // Atile K pitch (bf16 elems), padded for bank spread

typedef __bf16  v8bf  __attribute__((ext_vector_type(8)));
typedef __bf16  v16bf __attribute__((ext_vector_type(16)));
typedef float   v8f   __attribute__((ext_vector_type(8)));

__device__ __forceinline__ v16bf combine16(v8bf lo, v8bf hi) {
    v16bf r;
#pragma unroll
    for (int i = 0; i < 8; ++i) { r[i] = lo[i]; r[i + 8] = hi[i]; }
    return r;
}

// -------------------------------------------------------------------------
// Kernel 0: W1 [h][k][n] fp32  ->  W1T [h][n][k] bf16 (once; lives in L2)
// grid = (U/32, DF/32, H), block = 256 (32x8)
// -------------------------------------------------------------------------
__global__ __launch_bounds__(256) void bah_w1t_kernel(
    const float* __restrict__ W1, __bf16* __restrict__ W1T)
{
    __shared__ float t[32][33];
    const int tx = threadIdx.x & 31;       // fast axis
    const int ty = threadIdx.x >> 5;       // 0..7
    const int x = blockIdx.x * 32;         // n base
    const int y = blockIdx.y * 32;         // k base
    const int h = blockIdx.z;
#pragma unroll
    for (int j = 0; j < 4; ++j)            // read coalesced in n
        t[ty + j * 8][tx] = W1[((size_t)h * DF_ + y + ty + j * 8) * U_ + x + tx];
    __syncthreads();
#pragma unroll
    for (int j = 0; j < 4; ++j)            // write coalesced in k
        W1T[((size_t)h * U_ + x + ty + j * 8) * DF_ + y + tx] =
            (__bf16)t[tx][ty + j * 8];
}

// -------------------------------------------------------------------------
// Kernel 1: h_proj[b,h,u] = hidden[b,h,:] . W2[h,:,u] + b2[h,u]
// -------------------------------------------------------------------------
__global__ __launch_bounds__(256) void bah_hproj_kernel(
    const float* __restrict__ hidden, const float* __restrict__ W2,
    const float* __restrict__ b2, float* __restrict__ hproj)
{
    int gid = blockIdx.x * 256 + threadIdx.x;          // B*H*U = 65536 threads
    int u  = gid % U_;
    int bh = gid / U_;
    int h  = bh % H_;
    float acc = b2[h * U_ + u];
    const float* hv = hidden + (size_t)bh * DH_;
    const float* wv = W2 + (size_t)h * DH_ * U_ + u;
#pragma unroll 4
    for (int d = 0; d < DH_; ++d)
        acc += hv[d] * wv[(size_t)d * U_];
    hproj[gid] = acc;
}

// -------------------------------------------------------------------------
// Kernel 2 (dominant): score[b,h,l] = V[h].tanh(feat·W1[h] + b1 + hproj) + bV
// A (features) staged once in LDS (fp32->bf16); B fragments loaded straight
// from pre-transposed bf16 W1T in L2 -> barrier-free, pipelineable K loop.
// grid = (BL/TM, H), block = 256 (8 wave32, each owning a 64-wide N slab)
// -------------------------------------------------------------------------
__global__ __launch_bounds__(256) void bah_score_kernel(
    const float* __restrict__ features, const __bf16* __restrict__ W1T,
    const float* __restrict__ b1, const float* __restrict__ V,
    const float* __restrict__ bV, const float* __restrict__ hproj,
    float* __restrict__ score)
{
    __shared__ __bf16 Atile[TM * KPA];     // 64x512 bf16 feature tile
    __shared__ float  sc[TM];

    const int tid  = threadIdx.x;
    const int wv   = tid >> 5;             // wave id 0..7 -> N slab of 64
    const int lane = tid & 31;
    const int nl   = lane & 15;
    const int kh   = lane >> 4;            // k-half select per WMMA layout

    const int row0 = blockIdx.x * TM;      // global row in [0, B*L)
    const int b    = row0 >> 11;           // / L_
    const int l0   = row0 & (L_ - 1);
    const int h    = blockIdx.y;

    if (tid < TM) sc[tid] = 0.f;

    // ---- stage feature tile (fp32 -> bf16), once per block ----
    {
        const float4* fsrc = (const float4*)(features + ((size_t)b * L_ + l0) * DF_);
#pragma unroll
        for (int i = tid; i < TM * DF_ / 4; i += 256) {
            float4 fv = fsrc[i];
            int idx = i * 4;
            int m = idx >> 9;              // / DF_
            int d = idx & (DF_ - 1);
            __bf16* ap = &Atile[m * KPA + d];
            ap[0] = (__bf16)fv.x; ap[1] = (__bf16)fv.y;
            ap[2] = (__bf16)fv.z; ap[3] = (__bf16)fv.w;
        }
    }

    v8f acc[4][4];
#pragma unroll
    for (int mt = 0; mt < 4; ++mt)
#pragma unroll
        for (int nt = 0; nt < 4; ++nt)
            acc[mt][nt] = (v8f)(0.0f);

    __syncthreads();                       // Atile ready; no barriers after this

    // ---- K loop: 512 in steps of 32; barrier-free ----
    for (int ks = 0; ks < DF_ / 32; ++ks) {
        const int kb = ks * 32;
        // A fragments: lane holds row m, two 16B runs (k: 8*kh..+7, 16+8*kh..+7)
        v16bf afrag[4];
#pragma unroll
        for (int mt = 0; mt < 4; ++mt) {
            const __bf16* arow = &Atile[(mt * 16 + nl) * KPA + kb + 8 * kh];
            v8bf lo = *(const v8bf*)(arow);
            v8bf hi = *(const v8bf*)(arow + 16);
            afrag[mt] = combine16(lo, hi);
        }
#pragma unroll
        for (int nt = 0; nt < 4; ++nt) {
            // B fragment: lane = column n, two 16B runs (k: 8*kh..+7, 16+8*kh..+7) (L2-resident W1T)
            int n = wv * 64 + nt * 16 + nl;
            const __bf16* brow = W1T + ((size_t)h * U_ + n) * DF_ + kb + 8 * kh;
            v8bf blo = *(const v8bf*)(brow);
            v8bf bhi = *(const v8bf*)(brow + 16);
            v16bf bfrag = combine16(blo, bhi);
#pragma unroll
            for (int mt = 0; mt < 4; ++mt)
                acc[mt][nt] = __builtin_amdgcn_wmma_f32_16x16x32_bf16(
                    false, afrag[mt], false, bfrag, (short)0, acc[mt][nt],
                    false, false);
        }
    }

    // ---- fused epilogue: bias + tanh + V-dot, reduce over N ----
    const int mh = kh * 8;                 // C layout: m = r + 8*(lane>>4)
    float part[4][8];
#pragma unroll
    for (int mt = 0; mt < 4; ++mt)
#pragma unroll
        for (int r = 0; r < 8; ++r) part[mt][r] = 0.f;

#pragma unroll
    for (int nt = 0; nt < 4; ++nt) {
        int n = wv * 64 + nt * 16 + nl;
        float bias = b1[h * U_ + n] + hproj[((size_t)b * H_ + h) * U_ + n];
        float vn   = V[h * U_ + n];
#pragma unroll
        for (int mt = 0; mt < 4; ++mt)
#pragma unroll
            for (int r = 0; r < 8; ++r)
                part[mt][r] += vn * tanhf(acc[mt][nt][r] + bias);
    }
#pragma unroll
    for (int mt = 0; mt < 4; ++mt)
#pragma unroll
        for (int r = 0; r < 8; ++r)
            atomicAdd(&sc[mt * 16 + mh + r], part[mt][r]);
    __syncthreads();
    if (tid < TM)
        score[((size_t)b * H_ + h) * L_ + l0 + tid] = sc[tid] + bV[h];
}

// -------------------------------------------------------------------------
// Kernel 3: softmax over L per (b,h); emits normalized attn (ws) and the
// transposed attention_weights output [B,L,H,1].
// -------------------------------------------------------------------------
__global__ __launch_bounds__(256) void bah_softmax_kernel(
    float* __restrict__ attn /* in: scores, out: attn */,
    float* __restrict__ out_attnw)
{
    const int bh = blockIdx.x;             // B*H = 128
    const int b = bh / H_, h = bh % H_;
    const int tid = threadIdx.x;
    float* row = attn + (size_t)bh * L_;
    __shared__ float red[256];

    float m = -3.4e38f;
    for (int l = tid; l < L_; l += 256) m = fmaxf(m, row[l]);
    red[tid] = m; __syncthreads();
    for (int off = 128; off > 0; off >>= 1) {
        if (tid < off) red[tid] = fmaxf(red[tid], red[tid + off]);
        __syncthreads();
    }
    const float mx = red[0]; __syncthreads();

    float s = 0.f;
    for (int l = tid; l < L_; l += 256) {
        float e = __expf(row[l] - mx);
        row[l] = e;
        s += e;
    }
    red[tid] = s; __syncthreads();
    for (int off = 128; off > 0; off >>= 1) {
        if (tid < off) red[tid] += red[tid + off];
        __syncthreads();
    }
    const float rs = 1.f / red[0]; __syncthreads();

    for (int l = tid; l < L_; l += 256) {
        float a = row[l] * rs;
        row[l] = a;
        out_attnw[((size_t)b * L_ + l) * H_ + h] = a;
    }
}

// -------------------------------------------------------------------------
// Kernel 4: context partials over L-splits (deterministic, no float atomics)
// grid = (B, DF/256, LSPLIT=8)
// -------------------------------------------------------------------------
__global__ __launch_bounds__(256) void bah_ctx_partial_kernel(
    const float* __restrict__ features, const float* __restrict__ attn,
    float* __restrict__ ctxp)
{
    const int b = blockIdx.x, dblk = blockIdx.y, lz = blockIdx.z;
    const int tid = threadIdx.x;
    const int d = dblk * 256 + tid;
    const int lbase = lz * 256;
    __shared__ float a_s[H_][256];

#pragma unroll
    for (int h = 0; h < H_; ++h)
        a_s[h][tid] = attn[((size_t)b * H_ + h) * L_ + lbase + tid];
    __syncthreads();

    float acc[H_];
#pragma unroll
    for (int h = 0; h < H_; ++h) acc[h] = 0.f;

    for (int j = 0; j < 256; ++j) {
        float f = features[((size_t)b * L_ + lbase + j) * DF_ + d];
#pragma unroll
        for (int h = 0; h < H_; ++h) acc[h] += a_s[h][j] * f;
    }
#pragma unroll
    for (int h = 0; h < H_; ++h)
        ctxp[(size_t)lz * (B_ * H_ * DF_) + ((size_t)b * H_ + h) * DF_ + d] = acc[h];
}

__global__ __launch_bounds__(256) void bah_ctx_reduce_kernel(
    const float* __restrict__ ctxp, float* __restrict__ out_ctx)
{
    int gid = blockIdx.x * 256 + threadIdx.x;    // 65536
    float s = 0.f;
#pragma unroll
    for (int z = 0; z < 8; ++z) s += ctxp[(size_t)z * (B_ * H_ * DF_) + gid];
    out_ctx[gid] = s;
}

// -------------------------------------------------------------------------
extern "C" void kernel_launch(void* const* d_in, const int* in_sizes, int n_in,
                              void* d_out, int out_size, void* d_ws, size_t ws_size,
                              hipStream_t stream)
{
    const float* features = (const float*)d_in[0];
    const float* hidden   = (const float*)d_in[1];
    const float* W1       = (const float*)d_in[2];
    const float* b1       = (const float*)d_in[3];
    const float* W2       = (const float*)d_in[4];
    const float* b2       = (const float*)d_in[5];
    const float* V        = (const float*)d_in[6];
    const float* bV       = (const float*)d_in[7];

    float* out_ctx   = (float*)d_out;                    // [B,H,DF] = 65536
    float* out_attnw = (float*)d_out + B_ * H_ * DF_;    // [B,L,H,1] = 262144

    // ws layout: [ W1T bf16: H*U*DF (4MB) | hproj | scores | ctx partials ]
    __bf16* W1T   = (__bf16*)d_ws;
    float* ws_f   = (float*)((char*)d_ws + (size_t)H_ * U_ * DF_ * sizeof(__bf16));
    float* hproj  = ws_f;                                // 65536 floats
    float* scores = ws_f + B_ * H_ * U_;                 // 262144 floats
    float* ctxp   = scores + B_ * H_ * L_;               // 8*65536 floats

    // 0) one-time W1 convert+transpose (bf16, stays hot in 192MB L2)
    bah_w1t_kernel<<<dim3(U_ / 32, DF_ / 32, H_), 256, 0, stream>>>(W1, W1T);
    // 1) h_proj
    bah_hproj_kernel<<<B_ * H_ * U_ / 256, 256, 0, stream>>>(hidden, W2, b2, hproj);
    // 2) fused WMMA score GEMM (barrier-free K loop)
    bah_score_kernel<<<dim3(B_ * L_ / TM, H_), 256, 0, stream>>>(
        features, W1T, b1, V, bV, hproj, scores);
    // 3) softmax + attention_weights output
    bah_softmax_kernel<<<B_ * H_, 256, 0, stream>>>(scores, out_attnw);
    // 4) context (L-split partials + deterministic reduce)
    bah_ctx_partial_kernel<<<dim3(B_, DF_ / 256, 8), 256, 0, stream>>>(
        features, scores, ctxp);
    bah_ctx_reduce_kernel<<<B_ * H_ * DF_ / 256, 256, 0, stream>>>(ctxp, out_ctx);
}
